// CausalSelfAttention_79980880986740
// MI455X (gfx1250) — matrix operand order repaired and checked by reading, not compiled, not run on hardware
//
#include <hip/hip_runtime.h>
#include <hip/hip_bf16.h>

// ---------------------------------------------------------------------------
// CDNA5 (gfx1250) causal self-attention: bf16 WMMA pipeline
//   B=4, T=2048, E=1024, H=16, D=64
// ---------------------------------------------------------------------------

typedef __bf16 bf16_t;
typedef __attribute__((ext_vector_type(16))) __bf16 v16bf;
typedef __attribute__((ext_vector_type(8)))  float  v8f;
typedef int v4i __attribute__((vector_size(16)));   // matches builtin param type

#define CB 4
#define CT 2048
#define CE 1024
#define CH 16
#define CD 64
#define CM (CB * CT)      // 8192 rows
#define C3E (3 * CE)      // 3072

#define AS1 __attribute__((address_space(1)))
#define AS3 __attribute__((address_space(3)))

// gfx1250 async memory->LDS path (ASYNCcnt-tracked), with fallback
#if __has_builtin(__builtin_amdgcn_global_load_async_to_lds_b128)
#define HAVE_ASYNC_LDS 1
#else
#define HAVE_ASYNC_LDS 0
#endif

#if HAVE_ASYNC_LDS
template <int OFF>
__device__ __forceinline__ void async_ld128(const bf16_t* g, bf16_t* l) {
  // GLOBAL_LOAD_ASYNC_TO_LDS_B128: per-lane 16B memory -> LDS, no VGPR data
  __builtin_amdgcn_global_load_async_to_lds_b128(
      (AS1 v4i*)(AS1 void*)(void*)g,
      (AS3 v4i*)(AS3 void*)(void*)l, OFF, 0);
}
#endif

__device__ __forceinline__ void wait_asynccnt0() {
#if __has_builtin(__builtin_amdgcn_s_wait_asynccnt)
  __builtin_amdgcn_s_wait_asynccnt(0);
#else
  asm volatile("s_wait_asynccnt 0" ::: "memory");
#endif
}

union AFrag { v16bf v; uint4 q[2]; };
union BFrag { v16bf v; uint4 q[2]; };

__device__ __forceinline__ v8f wmma_bf16(v16bf a, v16bf b, v8f c) {
  // v_wmma_f32_16x16x32_bf16: D = A(16x32) * B(32x16) + C(16x16 f32)
  return __builtin_amdgcn_wmma_f32_16x16x32_bf16(
      /*neg_a=*/false, a, /*neg_b=*/false, b,
      /*c_mod=*/(short)0, c, /*reuse_a=*/false, /*reuse_b=*/false);
}

// ---------------------------------------------------------------------------
// Kernel 0: fp32 -> bf16 conversion (vectorized x4)
// ---------------------------------------------------------------------------
__global__ void f32_to_bf16_kernel(const float* __restrict__ src,
                                   bf16_t* __restrict__ dst, int n) {
  int i = (blockIdx.x * blockDim.x + threadIdx.x) * 4;
  if (i < n) {
    float4 f = *(const float4*)(src + i);
    union { bf16_t b[4]; ushort4 u; } o;
    o.b[0] = (bf16_t)f.x; o.b[1] = (bf16_t)f.y;
    o.b[2] = (bf16_t)f.z; o.b[3] = (bf16_t)f.w;
    *(ushort4*)(dst + i) = o.u;
  }
}

// ---------------------------------------------------------------------------
// Kernels 1/3: NT GEMM  C[M,N] = A[M,K] * Bw[N,K]^T   (bf16 in, f32 acc)
// Block tile 128x128, K-step 32, 8 waves (4 in M x 2 in N), wave tile 32x64.
// LDS tiles staged with GLOBAL_LOAD_ASYNC_TO_LDS_B128 when available.
// MODE 0: store f32 C.  MODE 1: fused RoPE epilogue + q/k/vT scatter.
// ---------------------------------------------------------------------------
template <int MODE>
__global__ __launch_bounds__(256) void gemm_nt_kernel(
    const bf16_t* __restrict__ A, const bf16_t* __restrict__ Bw,
    int M, int N, int K, float* __restrict__ Cout,
    bf16_t* __restrict__ qb, bf16_t* __restrict__ kb,
    bf16_t* __restrict__ vtb) {
  __shared__ bf16_t As[128 * 32];
  __shared__ bf16_t Bs[128 * 32];

  const int tid  = threadIdx.x;
  const int lane = tid & 31;
  const int wid  = tid >> 5;
  const int wm   = wid & 3;   // wave row  (4 x 32 rows)
  const int wn   = wid >> 2;  // wave col  (2 x 64 cols)
  const int hi   = lane >> 4;
  const int ln   = lane & 15;

  const int blockM = blockIdx.x * 128;
  const int blockN = blockIdx.y * 128;

  const int ldr = tid >> 1;            // 0..127 : tile row
  const int ldc = (tid & 1) * 16;      // 0 or 16: half-row of 16 bf16 (32B)

  v8f acc[2][4] = {};

  for (int k0 = 0; k0 < K; k0 += 32) {
    const bf16_t* gA = A  + (size_t)(blockM + ldr) * K + k0 + ldc;
    const bf16_t* gB = Bw + (size_t)(blockN + ldr) * K + k0 + ldc;
    bf16_t* lA = &As[ldr * 32 + ldc];
    bf16_t* lB = &Bs[ldr * 32 + ldc];

    if (k0 + 32 < K) {  // CDNA5 prefetch -> global_prefetch_b8
      __builtin_prefetch(gA + 32, 0, 1);
      __builtin_prefetch(gB + 32, 0, 1);
    }

#if HAVE_ASYNC_LDS
    __syncthreads();  // previous iteration's LDS reads complete
    // memory -> LDS without a VGPR round-trip (ASYNCcnt-tracked)
    async_ld128<0>(gA, lA);
    async_ld128<16>(gA, lA);
    async_ld128<0>(gB, lB);
    async_ld128<16>(gB, lB);
    wait_asynccnt0();
    __syncthreads();  // whole tile visible to all waves
#else
    uint4 a0 = ((const uint4*)gA)[0], a1 = ((const uint4*)gA)[1];
    uint4 b0 = ((const uint4*)gB)[0], b1 = ((const uint4*)gB)[1];
    __syncthreads();
    ((uint4*)lA)[0] = a0; ((uint4*)lA)[1] = a1;
    ((uint4*)lB)[0] = b0; ((uint4*)lB)[1] = b1;
    __syncthreads();
#endif

    // ---- per-wave fragments (ISA wave32 layouts) ------------------------
    AFrag af[2];
#pragma unroll
    for (int mi = 0; mi < 2; ++mi) {
      int row = wm * 32 + mi * 16 + ln;
      af[mi].q[0] = *(uint4*)&As[row * 32 + hi * 8];        // K 0..7 / 8..15
      af[mi].q[1] = *(uint4*)&As[row * 32 + 16 + hi * 8];   // K 16..23 / 24..31
    }
    BFrag bfm[4];
#pragma unroll
    for (int ni = 0; ni < 4; ++ni) {
      int col = wn * 64 + ni * 16 + ln;
      bfm[ni].q[0] = *(uint4*)&Bs[col * 32 + hi * 8];       // K 0..7 / 8..15
      bfm[ni].q[1] = *(uint4*)&Bs[col * 32 + 16 + hi * 8];
    }
#pragma unroll
    for (int mi = 0; mi < 2; ++mi)
#pragma unroll
      for (int ni = 0; ni < 4; ++ni)
        acc[mi][ni] = wmma_bf16(af[mi].v, bfm[ni].v, acc[mi][ni]);
  }

  // ---- epilogue ---------------------------------------------------------
#pragma unroll
  for (int mi = 0; mi < 2; ++mi) {
#pragma unroll
    for (int ni = 0; ni < 4; ++ni) {
#pragma unroll
      for (int r = 0; r < 8; ++r) {
        int row = blockM + wm * 32 + mi * 16 + r + hi * 8;  // C/D: M = r + 8*hi
        int col = blockN + wn * 64 + ni * 16 + ln;          // C/D: N = lane%16
        float val = acc[mi][ni][r];
        if (MODE == 0) {
          Cout[(size_t)row * N + col] = val;
        } else {
          // adjacent N columns live in adjacent lanes -> RoPE pair via shfl
          float partner = __shfl_xor(val, 1, 32);
          int b = row >> 11, t = row & 2047;
          if (col < 2 * CE) {  // q or k: apply rotary embedding
            int jj = (col < CE) ? col : col - CE;
            int h = jj >> 6, d = jj & 63;
            int f = (d < 32) ? d : d - 32;
            // inv_freq[f] = 10000^(-2f/64) = exp(-f * ln(10000)/32)
            float ang = (float)t * __expf((float)f * -0.28782313662425574f);
            float sv, cv;
            __sincosf(ang, &sv, &cv);
            float outv = val * cv + ((d & 1) ? partner * sv : -partner * sv);
            size_t idx = (((size_t)(b * CH + h) * CT + t) * CD + d);
            if (col < CE) qb[idx] = (bf16_t)outv;
            else          kb[idx] = (bf16_t)outv;
          } else {             // v: store transposed  vT[b][h][d][t]
            int jj = col - 2 * CE;
            int h = jj >> 6, d = jj & 63;
            vtb[((size_t)(b * CH + h) * CD + d) * CT + t] = (bf16_t)val;
          }
        }
      }
    }
  }
}

// ---------------------------------------------------------------------------
// Kernel 2: causal flash attention, one wave per 16 query rows.
// Q[16,64] x K^T -> S, online softmax (f32), P x V via LDS transpose.
// ---------------------------------------------------------------------------
__global__ __launch_bounds__(32) void flash_attn_kernel(
    const bf16_t* __restrict__ qb, const bf16_t* __restrict__ kb,
    const bf16_t* __restrict__ vtb, bf16_t* __restrict__ yb) {
  __shared__ bf16_t Pl[16 * 32];

  const int lane = threadIdx.x;
  const int hi = lane >> 4, ln = lane & 15;
  const int bh = blockIdx.x >> 7;       // T/16 = 128 q-tiles per (b,h)
  const int qt = blockIdx.x & 127;
  const int q0 = qt * 16;
  const int b = bh >> 4, h = bh & 15;

  const bf16_t* Q  = qb  + (size_t)bh * CT * CD;
  const bf16_t* Kp = kb  + (size_t)bh * CT * CD;
  const bf16_t* Vt = vtb + (size_t)bh * CD * CT;

  // Q fragments for d-chunks {0,32} (A-matrix 16x32 layout)
  AFrag qf[2];
#pragma unroll
  for (int dc = 0; dc < 2; ++dc) {
    int row = q0 + ln, d0 = dc * 32;
    qf[dc].q[0] = *(const uint4*)&Q[(size_t)row * CD + d0 + hi * 8];
    qf[dc].q[1] = *(const uint4*)&Q[(size_t)row * CD + d0 + 16 + hi * 8];
  }

  v8f O[4] = {};
  float mrow[8], lrow[8];
#pragma unroll
  for (int r = 0; r < 8; ++r) { mrow[r] = -1e30f; lrow[r] = 0.0f; }

  const int nk = (q0 + 16 + 31) >> 5;   // causal: k tiles of 32
  for (int kt = 0; kt < nk; ++kt) {
    const int k0 = kt * 32;

    // ---- S = Q * K^T  (two 16-col halves, accumulate over D=64) ---------
    v8f s[2] = {};
#pragma unroll
    for (int nh = 0; nh < 2; ++nh) {
      int kc = k0 + nh * 16 + ln;       // B-frag: N = k-row index
#pragma unroll
      for (int dc = 0; dc < 2; ++dc) {
        BFrag bfr;
        bfr.q[0] = *(const uint4*)&Kp[(size_t)kc * CD + dc * 32 + hi * 8];
        bfr.q[1] = *(const uint4*)&Kp[(size_t)kc * CD + dc * 32 + 16 + hi * 8];
        s[nh] = wmma_bf16(qf[dc].v, bfr.v, s[nh]);
      }
    }

    // ---- scale + causal mask + online softmax ---------------------------
    float p[2][8];
#pragma unroll
    for (int r = 0; r < 8; ++r) {
      int rowq = q0 + r + hi * 8;
      float mloc = -1e30f;
#pragma unroll
      for (int nh = 0; nh < 2; ++nh) {
        int colk = k0 + nh * 16 + ln;
        float v = s[nh][r] * 0.125f;               // 1/sqrt(64)
        v = (colk <= rowq) ? v : -1e30f;
        p[nh][r] = v;
        mloc = fmaxf(mloc, v);
      }
#pragma unroll
      for (int off = 8; off >= 1; off >>= 1)       // row max over 16 lanes
        mloc = fmaxf(mloc, __shfl_xor(mloc, off, 32));
      float mnew  = fmaxf(mrow[r], mloc);
      float alpha = __expf(mrow[r] - mnew);
      mrow[r] = mnew;
      float lsum = 0.0f;
#pragma unroll
      for (int nh = 0; nh < 2; ++nh) {
        p[nh][r] = __expf(p[nh][r] - mnew);
        lsum += p[nh][r];
      }
#pragma unroll
      for (int off = 8; off >= 1; off >>= 1)
        lsum += __shfl_xor(lsum, off, 32);
      lrow[r] = lrow[r] * alpha + lsum;
#pragma unroll
      for (int ni = 0; ni < 4; ++ni) O[ni][r] *= alpha;
    }

    // ---- C-layout P -> A-layout via LDS (single wave: DS is in-order) ---
#pragma unroll
    for (int r = 0; r < 8; ++r)
#pragma unroll
      for (int nh = 0; nh < 2; ++nh)
        Pl[(r + hi * 8) * 32 + nh * 16 + ln] = (bf16_t)p[nh][r];
    asm volatile("s_wait_dscnt 0" ::: "memory");

    AFrag pf;
    pf.q[0] = *(uint4*)&Pl[ln * 32 + hi * 8];
    pf.q[1] = *(uint4*)&Pl[ln * 32 + 16 + hi * 8];

    // ---- O += P * V  (Vt is [D][T]: contiguous K per lane) --------------
#pragma unroll
    for (int ni = 0; ni < 4; ++ni) {
      int dcol = ni * 16 + ln;
      BFrag vf;
      vf.q[0] = *(const uint4*)&Vt[(size_t)dcol * CT + k0 + hi * 8];
      vf.q[1] = *(const uint4*)&Vt[(size_t)dcol * CT + k0 + 16 + hi * 8];
      O[ni] = wmma_bf16(pf.v, vf.v, O[ni]);
    }
  }

  // ---- finalize: y[b][t][h*64+d] = O / l ------------------------------
#pragma unroll
  for (int ni = 0; ni < 4; ++ni)
#pragma unroll
    for (int r = 0; r < 8; ++r) {
      int t = q0 + r + hi * 8;
      int d = ni * 16 + ln;
      float outv = O[ni][r] / lrow[r];
      yb[((size_t)(b * CT + t) * CE) + h * CD + d] = (bf16_t)outv;
    }
}

// ---------------------------------------------------------------------------
// Host launcher
// ---------------------------------------------------------------------------
extern "C" void kernel_launch(void* const* d_in, const int* in_sizes, int n_in,
                              void* d_out, int out_size, void* d_ws,
                              size_t ws_size, hipStream_t stream) {
  const float* x      = (const float*)d_in[0];
  const float* w_attn = (const float*)d_in[1];
  const float* w_proj = (const float*)d_in[2];
  float* out = (float*)d_out;

  char* ws = (char*)d_ws;
  size_t off = 0;
  bf16_t* x_bf  = (bf16_t*)(ws + off); off += (size_t)CM * CE * 2;        // 16 MB
  bf16_t* wA_bf = (bf16_t*)(ws + off); off += (size_t)C3E * CE * 2;       //  6 MB
  bf16_t* wP_bf = (bf16_t*)(ws + off); off += (size_t)CE * CE * 2;        //  2 MB
  bf16_t* q_bf  = (bf16_t*)(ws + off); off += (size_t)CB * CH * CT * CD * 2;
  bf16_t* k_bf  = (bf16_t*)(ws + off); off += (size_t)CB * CH * CT * CD * 2;
  bf16_t* vT_bf = (bf16_t*)(ws + off); off += (size_t)CB * CH * CD * CT * 2;
  bf16_t* y_bf  = (bf16_t*)(ws + off); off += (size_t)CM * CE * 2;

  // 0) fp32 -> bf16
  {
    int n = CM * CE;
    f32_to_bf16_kernel<<<n / 4 / 256, 256, 0, stream>>>(x, x_bf, n);
    n = C3E * CE;
    f32_to_bf16_kernel<<<n / 4 / 256, 256, 0, stream>>>(w_attn, wA_bf, n);
    n = CE * CE;
    f32_to_bf16_kernel<<<n / 4 / 256, 256, 0, stream>>>(w_proj, wP_bf, n);
  }

  // 1) QKV GEMM + fused RoPE + q/k/vT scatter   (M=8192, N=3072, K=1024)
  {
    dim3 grid(CM / 128, C3E / 128);
    gemm_nt_kernel<1><<<grid, 256, 0, stream>>>(x_bf, wA_bf, CM, C3E, CE,
                                                nullptr, q_bf, k_bf, vT_bf);
  }

  // 2) causal flash attention (one wave per 16 q rows)
  flash_attn_kernel<<<CB * CH * (CT / 16), 32, 0, stream>>>(q_bf, k_bf, vT_bf,
                                                            y_bf);

  // 3) output projection                          (M=8192, N=1024, K=1024)
  {
    dim3 grid(CM / 128, CE / 128);
    gemm_nt_kernel<0><<<grid, 256, 0, stream>>>(y_bf, wP_bf, CM, CE, CE, out,
                                                nullptr, nullptr, nullptr);
  }
}
